// GLAttention_61469571940758
// MI455X (gfx1250) — compile-verified, matrix operand order checked by reading, not run on hardware
//
#include <hip/hip_runtime.h>

// ---------------- problem constants ----------------
#define BWIN  256          // number of windows (B_)
#define NTOK  343          // tokens per window
#define NPAD  352          // padded to 22 tiles of 16
#define CDIM  192
#define NHEAD 6
#define HD    32           // head dim (== one WMMA K step)
#define MT    22           // NPAD/16 row tiles
#define QKV_NT 36          // 576/16 col tiles
#define PRJ_NT 12          // 192/16 col tiles

typedef __attribute__((ext_vector_type(16))) __bf16       v16bf;
typedef __attribute__((ext_vector_type(8)))  float        v8f;
typedef __attribute__((ext_vector_type(4)))  float        f32x4;
typedef __attribute__((ext_vector_type(4)))  unsigned int u32x4;
typedef __attribute__((ext_vector_type(2)))  unsigned int u32x2;

union BF16x16 { u32x4 q[2]; v16bf v; };

// D = A(16x32 bf16) * B(32x16 bf16) + C(16x16 f32)
__device__ inline v8f wmma_bf16(v16bf a, v16bf b, v8f c) {
    return __builtin_amdgcn_wmma_f32_16x16x32_bf16(
        /*neg_a=*/false, a, /*neg_b=*/false, b,
        /*c_mod=*/(short)0, c, /*reuse_a=*/false, /*reuse_b=*/false);
}

// Load one lane's 16 bf16 elements of an A/B tile: two contiguous 16B chunks
// at p and p+16 elements (K-chunks {base..base+7} and {base+16..base+23}).
__device__ inline v16bf load_bf16_tile(const __bf16* p) {
    BF16x16 u;
    u.q[0] = *reinterpret_cast<const u32x4*>(p);
    u.q[1] = *reinterpret_cast<const u32x4*>(p + 16);
    return u.v;
}

// ---------------- pack kernels: f32 -> bf16, vectorized by 4 ----------------
__global__ __launch_bounds__(256) void pack_bf16_kernel(
        const float* __restrict__ src, __bf16* __restrict__ dst, int n4) {
    int i = blockIdx.x * blockDim.x + threadIdx.x;
    if (i < n4) {
        f32x4 v = *reinterpret_cast<const f32x4*>(src + (size_t)i * 4);
        union { u32x2 u; __bf16 h[4]; } P;
        P.h[0] = (__bf16)v.x; P.h[1] = (__bf16)v.y;
        P.h[2] = (__bf16)v.z; P.h[3] = (__bf16)v.w;
        *reinterpret_cast<u32x2*>(dst + (size_t)i * 4) = P.u;
    }
}

// x [BWIN][NTOK][CDIM] f32 -> xb16 [BWIN][NPAD][CDIM] bf16 (pad rows zeroed)
__global__ __launch_bounds__(256) void pack_x_kernel(
        const float* __restrict__ x, __bf16* __restrict__ xb16) {
    const int C4 = CDIM / 4;
    int i = blockIdx.x * blockDim.x + threadIdx.x;   // over BWIN*NPAD*C4
    if (i >= BWIN * NPAD * C4) return;
    int c4  = i % C4;
    int row = (i / C4) % NPAD;
    int b   = i / (C4 * NPAD);
    union { u32x2 u; __bf16 h[4]; } P;
    if (row < NTOK) {
        f32x4 v = *reinterpret_cast<const f32x4*>(
            x + ((size_t)(b * NTOK + row)) * CDIM + c4 * 4);
        P.h[0] = (__bf16)v.x; P.h[1] = (__bf16)v.y;
        P.h[2] = (__bf16)v.z; P.h[3] = (__bf16)v.w;
    } else {
        P.h[0] = P.h[1] = P.h[2] = P.h[3] = (__bf16)0.f;
    }
    *reinterpret_cast<u32x2*>(xb16 + (size_t)i * 4) = P.u;
}

// ---------------- bias expand (transposed, q-padded):
// biasT[h][col][q] = bias_table[rel_index[q,col]][h], q padded to NPAD (zeros)
__global__ __launch_bounds__(256) void bias_expand_kernel(
        const float* __restrict__ bias_table, const int* __restrict__ rel_index,
        float* __restrict__ biasT) {
    int i = blockIdx.x * blockDim.x + threadIdx.x;   // over NTOK(col) * NPAD(q)
    if (i >= NTOK * NPAD) return;
    int q   = i % NPAD;
    int col = i / NPAD;
    if (q < NTOK) {
        int idx = rel_index[q * NTOK + col];
        #pragma unroll
        for (int h = 0; h < NHEAD; ++h)
            biasT[((size_t)(h * NTOK) + col) * NPAD + q] = bias_table[idx * NHEAD + h];
    } else {
        #pragma unroll
        for (int h = 0; h < NHEAD; ++h)
            biasT[((size_t)(h * NTOK) + col) * NPAD + q] = 0.f;
    }
}

// ---------------- kernel 1: qkv = x @ w_qkv^T + b_qkv (all-bf16 operands),
// A fragments cached in registers per row-tile; Q scaled, V stored d-major.
// Pad rows (343..351) are stored unconditionally: xb16 pads are zero so they
// hold the bias value; K pads are masked in attention, V pads meet P==0.
__global__ __launch_bounds__(256) void qkv_kernel(
        const __bf16* __restrict__ xb16, const __bf16* __restrict__ wq16,
        const float* __restrict__ b_qkv,
        __bf16* __restrict__ qb, __bf16* __restrict__ kbuf, __bf16* __restrict__ vt) {
    const int b    = blockIdx.x;
    const int tid  = threadIdx.x;
    const int wave = tid >> 5, lane = tid & 31;
    const int half = lane >> 4, l16 = lane & 15;
    const float scale = 0.17677669529663687f;   // 1/sqrt(32)

    const __bf16* xb = xb16 + (size_t)b * NPAD * CDIM;
    for (int mt = wave; mt < MT; mt += 8) {
        // cache A fragments (row = l16 within tile) for all 6 k-steps
        const __bf16* xr = xb + (size_t)(mt * 16 + l16) * CDIM;
        v16bf A[6];
        #pragma unroll
        for (int kk = 0; kk < 6; ++kk)
            A[kk] = load_bf16_tile(xr + kk * 32 + half * 8);

        for (int nt = 0; nt < QKV_NT; ++nt) {
            int o = nt * 16 + l16;                  // B-lane output column
            const __bf16* wr = wq16 + (size_t)o * CDIM;
            v8f acc = {};
            #pragma unroll
            for (int kk = 0; kk < 6; ++kk)
                acc = wmma_bf16(A[kk], load_bf16_tile(wr + kk * 32 + half * 8), acc);

            // epilogue: C layout row = r + 8*half, col = o (lane); sec/h/d are
            // wave-uniform per nt -> uniform branches, unconditional stores
            int sec = nt / PRJ_NT;                  // 0=q 1=k 2=v
            int oo  = (nt % PRJ_NT) * 16;           // base within 192
            int h   = oo / HD;
            int d   = (oo % HD) + l16;
            float bias = b_qkv[o];
            size_t bh  = (size_t)(b * NHEAD + h);
            int row0   = mt * 16 + 8 * half;
            if (sec == 0) {
                __bf16* dst = qb + (bh * NPAD + row0) * HD + d;
                #pragma unroll
                for (int r = 0; r < 8; ++r) dst[r * HD] = (__bf16)((acc[r] + bias) * scale);
            } else if (sec == 1) {
                __bf16* dst = kbuf + (bh * NPAD + row0) * HD + d;
                #pragma unroll
                for (int r = 0; r < 8; ++r) dst[r * HD] = (__bf16)(acc[r] + bias);
            } else {
                __bf16* dst = vt + (bh * HD + d) * NPAD + row0;
                #pragma unroll
                for (int r = 0; r < 8; ++r) dst[r] = (__bf16)(acc[r] + bias);
            }
        }
    }
}

// ---------------- kernel 2: flash attention, one wave per (b,h,q-tile of 16 rows)
__global__ __launch_bounds__(32) void attn_kernel(
        const __bf16* __restrict__ qb, const __bf16* __restrict__ kbuf,
        const __bf16* __restrict__ vt, const float* __restrict__ biasT,
        __bf16* __restrict__ ao) {
    __shared__ __align__(16) __bf16 pls[16 * 32];   // P tile staging (transpose)

    const int gid = blockIdx.x;                 // = bh*22 + qt
    const int qt  = gid % MT;
    const int bh  = gid / MT;
    const int h   = bh % NHEAD;
    const int bb  = bh / NHEAD;
    const int lane = threadIdx.x, half = lane >> 4, l16 = lane & 15;

    const __bf16* qbase = qb   + (size_t)bh * NPAD * HD;
    const __bf16* kbase = kbuf + (size_t)bh * NPAD * HD;
    const __bf16* vbase = vt   + (size_t)bh * HD * NPAD;
    const float*  bhp   = biasT + (size_t)(h * NTOK) * NPAD;
    const int     qoff  = qt * 16 + 8 * half;   // first of this lane's 8 C rows

    // Q A-tile: lane holds row l16, k-chunks {half*8..}, {16+half*8..}
    v16bf Aq = load_bf16_tile(qbase + (size_t)(qt * 16 + l16) * HD + half * 8);

    v8f olo = {}, ohi = {};
    float m[8], l[8];
    #pragma unroll
    for (int r = 0; r < 8; ++r) { m[r] = -3.0e38f; l[r] = 0.f; }

    for (int kp = 0; kp < 11; ++kp) {           // 32 key positions per step
        if (kp < 10) {
            __builtin_prefetch(kbase + (size_t)((kp + 1) * 32 + l16) * HD, 0, 1);
            __builtin_prefetch(vbase + (size_t)l16 * NPAD + (kp + 1) * 32, 0, 1);
        }

        v8f s[2];
        #pragma unroll
        for (int t = 0; t < 2; ++t) {
            int kt = kp * 2 + t;
            // B tile of K^T: lane holds key-token (kt*16+l16), d-chunks like A
            v16bf Bk = load_bf16_tile(kbase + (size_t)(kt * 16 + l16) * HD + half * 8);
            v8f z = {};
            s[t] = wmma_bf16(Aq, Bk, z);
            int col = kt * 16 + l16;
            if (col < NTOK) {
                // per-lane fixed col; 8 consecutive q values -> two b128 loads
                const float* bp = bhp + (size_t)col * NPAD + qoff;
                f32x4 b0 = *reinterpret_cast<const f32x4*>(bp);
                f32x4 b1 = *reinterpret_cast<const f32x4*>(bp + 4);
                s[t][0] += b0.x; s[t][1] += b0.y; s[t][2] += b0.z; s[t][3] += b0.w;
                s[t][4] += b1.x; s[t][5] += b1.y; s[t][6] += b1.z; s[t][7] += b1.w;
            } else {
                #pragma unroll
                for (int r = 0; r < 8; ++r) s[t][r] = -1.0e30f;   // mask pad keys
            }
        }
        // online softmax over these 32 columns; pack P into LDS (bf16)
        #pragma unroll
        for (int r = 0; r < 8; ++r) {
            float rv = fmaxf(s[0][r], s[1][r]);
            rv = fmaxf(rv, __shfl_xor(rv, 1, 32));
            rv = fmaxf(rv, __shfl_xor(rv, 2, 32));
            rv = fmaxf(rv, __shfl_xor(rv, 4, 32));
            rv = fmaxf(rv, __shfl_xor(rv, 8, 32));
            float mn   = fmaxf(m[r], rv);
            float corr = __expf(m[r] - mn);
            m[r] = mn;
            float p0 = __expf(s[0][r] - mn);
            float p1 = __expf(s[1][r] - mn);
            float ps = p0 + p1;
            ps += __shfl_xor(ps, 1, 32);
            ps += __shfl_xor(ps, 2, 32);
            ps += __shfl_xor(ps, 4, 32);
            ps += __shfl_xor(ps, 8, 32);
            l[r] = l[r] * corr + ps;
            olo[r] *= corr;  ohi[r] *= corr;
            int prow = r + 8 * half;            // C-layout row owned by this lane
            pls[prow * 32 + l16]      = (__bf16)p0;
            pls[prow * 32 + 16 + l16] = (__bf16)p1;
        }
        __syncthreads();
        // re-read P in A layout (row = l16, k-chunks at half*8 and +16)
        v16bf Ap = load_bf16_tile(&pls[l16 * 32 + half * 8]);
        __syncthreads();
        // V B-tiles: lane holds output d = dt*16+l16, keypos-chunks contiguous (Vt is d-major)
        v16bf Bv0 = load_bf16_tile(vbase + (size_t)(l16)      * NPAD + kp * 32 + half * 8);
        v16bf Bv1 = load_bf16_tile(vbase + (size_t)(16 + l16) * NPAD + kp * 32 + half * 8);
        olo = wmma_bf16(Ap, Bv0, olo);
        ohi = wmma_bf16(Ap, Bv1, ohi);
    }
    // normalize and store to AO [b][NPAD][CDIM] bf16; pad rows stored too
    // (finite values; consumed only by kernel-3 pad rows which are never stored)
    {
        size_t base0 = ((size_t)bb * NPAD + qoff) * CDIM + h * HD;
        #pragma unroll
        for (int r = 0; r < 8; ++r) {
            float inv = (l[r] > 0.f) ? 1.0f / l[r] : 0.f;
            ao[base0 + (size_t)r * CDIM + l16]      = (__bf16)(olo[r] * inv);
            ao[base0 + (size_t)r * CDIM + 16 + l16] = (__bf16)(ohi[r] * inv);
        }
    }
}

// ---------------- kernel 3: out = AO @ w_proj^T + b_proj (fp32 out),
// A fragments cached in registers per row-tile; only the last row-tile is
// partial, handled by a wave-uniform branch.
__global__ __launch_bounds__(256) void proj_kernel(
        const __bf16* __restrict__ ao, const __bf16* __restrict__ wp16,
        const float* __restrict__ b_proj, float* __restrict__ out) {
    const int b    = blockIdx.x;
    const int tid  = threadIdx.x;
    const int wave = tid >> 5, lane = tid & 31;
    const int half = lane >> 4, l16 = lane & 15;
    const __bf16* aob = ao + (size_t)b * NPAD * CDIM;

    for (int mt = wave; mt < MT; mt += 8) {
        const __bf16* ar = aob + (size_t)(mt * 16 + l16) * CDIM;
        v16bf A[6];
        #pragma unroll
        for (int kk = 0; kk < 6; ++kk)
            A[kk] = load_bf16_tile(ar + kk * 32 + half * 8);

        const bool full = (mt * 16 + 16) <= NTOK;   // wave-uniform
        for (int nt = 0; nt < PRJ_NT; ++nt) {
            int o = nt * 16 + l16;
            const __bf16* wr = wp16 + (size_t)o * CDIM;
            v8f acc = {};
            #pragma unroll
            for (int kk = 0; kk < 6; ++kk)
                acc = wmma_bf16(A[kk], load_bf16_tile(wr + kk * 32 + half * 8), acc);

            float bias = b_proj[o];
            int row0 = mt * 16 + 8 * half;
            float* dst = out + ((size_t)b * NTOK + row0) * CDIM + o;
            if (full) {
                #pragma unroll
                for (int r = 0; r < 8; ++r)
                    dst[(size_t)r * CDIM] = acc[r] + bias;
            } else {
                #pragma unroll
                for (int r = 0; r < 8; ++r)
                    if (row0 + r < NTOK)
                        dst[(size_t)r * CDIM] = acc[r] + bias;
            }
        }
    }
}

// ---------------- host launcher ----------------
extern "C" void kernel_launch(void* const* d_in, const int* in_sizes, int n_in,
                              void* d_out, int out_size, void* d_ws, size_t ws_size,
                              hipStream_t stream) {
    (void)in_sizes; (void)n_in; (void)out_size; (void)ws_size;
    const float* x          = (const float*)d_in[0];
    const float* w_qkv      = (const float*)d_in[1];
    const float* b_qkv      = (const float*)d_in[2];
    const float* w_proj     = (const float*)d_in[3];
    const float* b_proj     = (const float*)d_in[4];
    const float* bias_table = (const float*)d_in[5];
    const int*   rel_index  = (const int*)d_in[6];
    // d_in[7] = is_global (==0, local window path)

    char* w = (char*)d_ws;
    const size_t PERB = (size_t)BWIN * NHEAD * NPAD * HD * 2;   // 34,603,008 B
    __bf16* qb    = (__bf16*)(w);
    __bf16* kbuf  = (__bf16*)(w + PERB);
    __bf16* vt    = (__bf16*)(w + 2 * PERB);
    __bf16* ao    = (__bf16*)(w + 3 * PERB);    // [BWIN][NPAD][CDIM], same size
    __bf16* xb16  = (__bf16*)(w + 4 * PERB);    // [BWIN][NPAD][CDIM], same size
    __bf16* wq16  = (__bf16*)(w + 5 * PERB);                       // 576*192 bf16
    __bf16* wp16  = (__bf16*)(w + 5 * PERB + (size_t)576 * 192 * 2); // 192*192 bf16
    float*  biasT = (float*) (w + 5 * PERB + (size_t)576 * 192 * 2
                                           + (size_t)192 * 192 * 2); // [H][343][352] f32

    const int xq4 = BWIN * NPAD * (CDIM / 4);
    pack_x_kernel<<<dim3((xq4 + 255) / 256), dim3(256), 0, stream>>>(x, xb16);
    pack_bf16_kernel<<<dim3((576 * 192 / 4 + 255) / 256), dim3(256), 0, stream>>>(
        w_qkv, wq16, 576 * 192 / 4);
    pack_bf16_kernel<<<dim3((192 * 192 / 4 + 255) / 256), dim3(256), 0, stream>>>(
        w_proj, wp16, 192 * 192 / 4);
    bias_expand_kernel<<<dim3((NTOK * NPAD + 255) / 256), dim3(256), 0, stream>>>(
        bias_table, rel_index, biasT);

    qkv_kernel<<<dim3(BWIN), dim3(256), 0, stream>>>(
        xb16, wq16, b_qkv, qb, kbuf, vt);
    attn_kernel<<<dim3(BWIN * NHEAD * MT), dim3(32), 0, stream>>>(
        qb, kbuf, vt, biasT, ao);
    proj_kernel<<<dim3(BWIN), dim3(256), 0, stream>>>(
        ao, wp16, b_proj, (float*)d_out);
}
